// SlotAttention_10041633538391
// MI455X (gfx1250) — matrix operand order repaired and checked by reading, not compiled, not run on hardware
//
#include <hip/hip_runtime.h>
#include <hip/hip_bf16.h>

// ---------------------------------------------------------------------------
// SlotAttention forward for MI455X (gfx1250, wave32).
// All 64-deep GEMMs use v_wmma_f32_16x16x32_f16 (f16 operands, f32 accum).
// LayerNorm is done in WMMA C-register layout with 16-lane shfl_xor reductions.
// Workspace requirement: ~144 MB (fits MI455X 192MB L2 for the hot tensors).
// ---------------------------------------------------------------------------

typedef __attribute__((ext_vector_type(16))) _Float16 v16h;
typedef __attribute__((ext_vector_type(8)))  float    v8f;

namespace sa {
constexpr int B  = 16;
constexpr int N  = 4096;   // H*W = 64*64
constexpr int D  = 64;
constexpr int SD = 64;
constexpr int K  = 5;
constexpr int CD = 8;
constexpr int ITERS = 4;
constexpr float SCALE = 0.125f;   // SD^-0.5
constexpr float EPS   = 1e-8f;
}

#define DEV __device__ __forceinline__

DEV int crow(int r){ return r + ((threadIdx.x & 31) >> 4) * 8; }   // C-layout row
DEV int ccol(int j){ return j * 16 + (threadIdx.x & 15); }         // C-layout col

DEV float rsum16(float v){            // sum within 16-lane half group
  v += __shfl_xor(v, 1, 32);
  v += __shfl_xor(v, 2, 32);
  v += __shfl_xor(v, 4, 32);
  v += __shfl_xor(v, 8, 32);
  return v;
}

DEV v8f vzero(){
  v8f z;
  #pragma unroll
  for (int i = 0; i < 8; ++i) z[i] = 0.f;
  return z;
}

// Load A fragments (16x64 f16, row-major, stride ld) -> two 16x32 A tiles.
DEV void load_A(const _Float16* S, int ld, v16h A[2]){
  int m = threadIdx.x & 15;
  int khalf = ((threadIdx.x & 31) >> 4) * 8;
  #pragma unroll
  for (int kt = 0; kt < 2; ++kt){
    v16h a;
    #pragma unroll
    for (int h = 0; h < 16; ++h){
      int k = kt * 32 + ((h < 8) ? 0 : 16) + khalf + (h & 7);
      a[h] = S[m * ld + k];
    }
    A[kt] = a;
  }
}

// B fragments for  out = X @ W^T,  W row-major (64 out, 64 in), f32 source.
DEV void load_WT(const float* W, v16h Bw[2][4]){
  int n = threadIdx.x & 15;
  int koff = ((threadIdx.x & 31) >> 4) * 8;
  #pragma unroll
  for (int kt = 0; kt < 2; ++kt)
    #pragma unroll
    for (int j = 0; j < 4; ++j){
      v16h bf;
      #pragma unroll
      for (int h = 0; h < 16; ++h)
        bf[h] = (_Float16)W[(j * 16 + n) * 64 + kt * 32 + ((h < 8) ? 0 : 16) + koff + (h & 7)];
      Bw[kt][j] = bf;
    }
}

// 16x64 = 16x64 @ 64x64 : 8 WMMA ops, accumulate into C (C-layout, 4 tiles).
DEV void mm16x64(const v16h A[2], const v16h Bw[2][4], v8f C[4]){
  #pragma unroll
  for (int j = 0; j < 4; ++j){
    C[j] = __builtin_amdgcn_wmma_f32_16x16x32_f16(false, A[0], false, Bw[0][j],
                                                  (short)0, C[j], false, false);
    C[j] = __builtin_amdgcn_wmma_f32_16x16x32_f16(false, A[1], false, Bw[1][j],
                                                  (short)0, C[j], false, false);
  }
}

DEV void load_Ctile(const float* src, v8f C[4]){   // row-major 16x64, stride 64
  #pragma unroll
  for (int r = 0; r < 8; ++r){ int row = crow(r);
    #pragma unroll
    for (int j = 0; j < 4; ++j) C[j][r] = src[row * 64 + ccol(j)];
  }
}

DEV void store_Ctile(const v8f C[4], float* dst){
  #pragma unroll
  for (int r = 0; r < 8; ++r){ int row = crow(r);
    #pragma unroll
    for (int j = 0; j < 4; ++j) dst[row * 64 + ccol(j)] = C[j][r];
  }
}

// LayerNorm over the 64-dim rows of a C-layout tile; emit f16 row-major to LDS.
DEV void ln_c_to_lds(const v8f C[4], const float lg[4], const float lb[4],
                     _Float16* S){
  #pragma unroll
  for (int r = 0; r < 8; ++r){
    float s = 0.f, s2 = 0.f;
    #pragma unroll
    for (int j = 0; j < 4; ++j){ float v = C[j][r]; s += v; s2 += v * v; }
    s = rsum16(s); s2 = rsum16(s2);
    float mean = s * (1.f / 64.f);
    float var  = s2 * (1.f / 64.f) - mean * mean;
    float inv  = rsqrtf(var + 1e-5f);
    int row = crow(r);
    #pragma unroll
    for (int j = 0; j < 4; ++j)
      S[row * 64 + ccol(j)] = (_Float16)((C[j][r] - mean) * inv * lg[j] + lb[j]);
  }
}

// ---------------------------------------------------------------------------
// Kernel 1: LN(feat) -> f16 ws ; LN(feat_color) -> f32 ws.
// ---------------------------------------------------------------------------
__global__ __launch_bounds__(256) void prep_ln(
    const float* __restrict__ feat, const float* __restrict__ g,
    const float* __restrict__ b,
    const float* __restrict__ fc, const float* __restrict__ gc,
    const float* __restrict__ bc,
    _Float16* __restrict__ featln, float* __restrict__ fcln)
{
  using namespace sa;
  int wid  = (blockIdx.x * blockDim.x + threadIdx.x) >> 5;
  int nwav = (gridDim.x * blockDim.x) >> 5;
  int lane = threadIdx.x & 31;
  for (int row = wid; row < B * N; row += nwav){
    const float* src = feat + (size_t)row * 64;
    float v0 = src[lane], v1 = src[lane + 32];
    float s = v0 + v1, s2 = v0 * v0 + v1 * v1;
    #pragma unroll
    for (int m = 1; m < 32; m <<= 1){ s += __shfl_xor(s, m, 32); s2 += __shfl_xor(s2, m, 32); }
    float mean = s * (1.f / 64.f);
    float var  = s2 * (1.f / 64.f) - mean * mean;
    float inv  = rsqrtf(var + 1e-5f);
    featln[(size_t)row * 64 + lane]      = (_Float16)((v0 - mean) * inv * g[lane] + b[lane]);
    featln[(size_t)row * 64 + lane + 32] = (_Float16)((v1 - mean) * inv * g[lane + 32] + b[lane + 32]);
  }
  for (int row = wid; row < B * N; row += nwav){
    if (lane < CD){
      float v = fc[(size_t)row * CD + lane];
      float s = v, s2 = v * v;
      #pragma unroll
      for (int m = 1; m < CD; m <<= 1){ s += __shfl_xor(s, m, 32); s2 += __shfl_xor(s2, m, 32); }
      float mean = s * (1.f / CD);
      float var  = s2 * (1.f / CD) - mean * mean;
      float inv  = rsqrtf(var + 1e-5f);
      fcln[(size_t)row * CD + lane] = (v - mean) * inv * gc[lane] + bc[lane];
    }
  }
}

// ---------------------------------------------------------------------------
// Kernel 2 (once): kf/vf = featln @ w_{k,v}fg^T ;
// k_bg/v_bg = LN(featln @ w_{k,v}bg^T + ge_bg) @ w_mlp_bg^T + b_mlp_bg.
// One wave handles a 16-row tile of (B*N). 48 WMMA per tile.
// ---------------------------------------------------------------------------
__global__ __launch_bounds__(32) void proj_static(
    const _Float16* __restrict__ featln,
    const float* __restrict__ w_kfg, const float* __restrict__ w_vfg,
    const float* __restrict__ w_kbg, const float* __restrict__ w_vbg,
    const float* __restrict__ w_grid, const float* __restrict__ b_grid,
    const float* __restrict__ ln_bg_g, const float* __restrict__ ln_bg_b,
    const float* __restrict__ w_mlp_bg, const float* __restrict__ b_mlp_bg,
    float* __restrict__ kf, float* __restrict__ vf,
    float* __restrict__ kbg, float* __restrict__ vbg)
{
  using namespace sa;
  __shared__ _Float16 S[16 * 64];
  const size_t r0 = (size_t)blockIdx.x * 16;

  float lg[4], lb[4], bm[4], g0[4], g1[4], gb[4];
  #pragma unroll
  for (int j = 0; j < 4; ++j){
    int col = ccol(j);
    lg[j] = ln_bg_g[col]; lb[j] = ln_bg_b[col];
    bm[j] = b_mlp_bg[col];
    g0[j] = w_grid[col * 4 + 0] - w_grid[col * 4 + 2];
    g1[j] = w_grid[col * 4 + 1] - w_grid[col * 4 + 3];
    gb[j] = b_grid[col];
  }
  float gx[8], gy[8];
  #pragma unroll
  for (int r = 0; r < 8; ++r){
    int n = (int)((r0 + crow(r)) & (N - 1));
    gx[r] = -1.f + (2.f / 63.f) * (float)(n & 63);
    gy[r] = -1.f + (2.f / 63.f) * (float)(n >> 6);
  }

  v16h A[2]; load_A(featln + r0 * 64, 64, A);
  v16h Bw[2][4]; v8f C[4];

  // ---- kf, vf ----
  load_WT(w_kfg, Bw);
  #pragma unroll
  for (int j = 0; j < 4; ++j) C[j] = vzero();
  mm16x64(A, Bw, C);  store_Ctile(C, kf + r0 * 64);

  load_WT(w_vfg, Bw);
  #pragma unroll
  for (int j = 0; j < 4; ++j) C[j] = vzero();
  mm16x64(A, Bw, C);  store_Ctile(C, vf + r0 * 64);

  // ---- k_bg ----
  load_WT(w_kbg, Bw);
  #pragma unroll
  for (int j = 0; j < 4; ++j) C[j] = vzero();
  mm16x64(A, Bw, C);
  #pragma unroll
  for (int r = 0; r < 8; ++r)
    #pragma unroll
    for (int j = 0; j < 4; ++j) C[j][r] += g0[j] * gx[r] + g1[j] * gy[r] + gb[j];
  ln_c_to_lds(C, lg, lb, S);
  __syncthreads();
  v16h A2[2]; load_A(S, 64, A2);
  load_WT(w_mlp_bg, Bw);
  #pragma unroll
  for (int j = 0; j < 4; ++j) C[j] = vzero();
  mm16x64(A2, Bw, C);
  #pragma unroll
  for (int r = 0; r < 8; ++r)
    #pragma unroll
    for (int j = 0; j < 4; ++j) C[j][r] += bm[j];
  store_Ctile(C, kbg + r0 * 64);
  __syncthreads();

  // ---- v_bg ----
  load_WT(w_vbg, Bw);
  #pragma unroll
  for (int j = 0; j < 4; ++j) C[j] = vzero();
  mm16x64(A, Bw, C);
  #pragma unroll
  for (int r = 0; r < 8; ++r)
    #pragma unroll
    for (int j = 0; j < 4; ++j) C[j][r] += g0[j] * gx[r] + g1[j] * gy[r] + gb[j];
  ln_c_to_lds(C, lg, lb, S);
  __syncthreads();
  load_A(S, 64, A2);
  load_WT(w_mlp_bg, Bw);
  #pragma unroll
  for (int j = 0; j < 4; ++j) C[j] = vzero();
  mm16x64(A2, Bw, C);
  #pragma unroll
  for (int r = 0; r < 8; ++r)
    #pragma unroll
    for (int j = 0; j < 4; ++j) C[j][r] += bm[j];
  store_Ctile(C, vbg + r0 * 64);
}

// ---------------------------------------------------------------------------
// Kernel 3: initialize slots / positions (broadcast over batch).
// ---------------------------------------------------------------------------
__global__ __launch_bounds__(256) void init_state(
    const float* __restrict__ s_fg0, const float* __restrict__ s_bg0,
    const float* __restrict__ pos0,
    float* __restrict__ slot_fg, float* __restrict__ slot_bg, float* __restrict__ pos)
{
  using namespace sa;
  int t = blockIdx.x * blockDim.x + threadIdx.x;
  if (t < B * 4 * SD) slot_fg[t] = s_fg0[t & (4 * SD - 1)];
  if (t < B * SD)     slot_bg[t] = s_bg0[t & (SD - 1)];
  if (t < B * 4 * 2)  pos[t]     = pos0[t & 7];
}

// ---------------------------------------------------------------------------
// Kernel 4 (per iter): q_fg = LN(slot_fg)@w_q^T ; q_bg = LN(slot_bg)@w_qbg^T.
// One 64-thread block per row; rows 0..63 = fg, 64..79 = bg.
// ---------------------------------------------------------------------------
__global__ __launch_bounds__(64) void q_proj(
    const float* __restrict__ slot_fg, const float* __restrict__ slot_bg,
    const float* __restrict__ lnq_g, const float* __restrict__ lnq_b,
    const float* __restrict__ w_q,
    const float* __restrict__ lnqb_g, const float* __restrict__ lnqb_b,
    const float* __restrict__ w_qbg,
    float* __restrict__ q_fg, float* __restrict__ q_bg)
{
  using namespace sa;
  __shared__ float shx[64], shl[64];
  int row = blockIdx.x;
  bool fg = row < B * 4;
  const float* x  = fg ? slot_fg + row * 64 : slot_bg + (row - B * 4) * 64;
  const float* g  = fg ? lnq_g : lnqb_g;
  const float* bb = fg ? lnq_b : lnqb_b;
  const float* W  = fg ? w_q : w_qbg;
  float* qo = fg ? q_fg + row * 64 : q_bg + (row - B * 4) * 64;
  int j = threadIdx.x;
  shx[j] = x[j];
  __syncthreads();
  float s = 0.f, s2 = 0.f;
  for (int d = 0; d < 64; ++d){ float v = shx[d]; s += v; s2 += v * v; }
  float mean = s * (1.f / 64.f);
  float var  = s2 * (1.f / 64.f) - mean * mean;
  float inv  = rsqrtf(var + 1e-5f);
  shl[j] = (shx[j] - mean) * inv * g[j] + bb[j];
  __syncthreads();
  float acc = 0.f;
  for (int d = 0; d < 64; ++d) acc += shl[d] * W[j * 64 + d];
  qo[j] = acc;
}

// ---------------------------------------------------------------------------
// Kernel 5 (per iter, the hot one): for each (b, slot, 16-token tile):
//   ge  = geC0*relx + geC1*rely + b_grid            (in C-layout registers)
//   k   = LN(kf + ge) @ w_mlp_fg^T + b_mlp_fg       (WMMA, f16 LDS staging)
//   logit[b,1+k,n] = (q . k_row) * SCALE            (shuffle reduce)
//   v   = LN(vf + ge) @ w_mlp_fg^T + b_mlp_fg  -> ws (for the slot update)
// ---------------------------------------------------------------------------
__global__ __launch_bounds__(32) void fg_kv_logits(
    const float* __restrict__ kf, const float* __restrict__ vf,
    const float* __restrict__ pos,
    const float* __restrict__ w_grid, const float* __restrict__ b_grid,
    const float* __restrict__ ln_fg_g, const float* __restrict__ ln_fg_b,
    const float* __restrict__ w_mlp_fg, const float* __restrict__ b_mlp_fg,
    const float* __restrict__ q_fg,
    float* __restrict__ vout, float* __restrict__ logits)
{
  using namespace sa;
  __shared__ _Float16 S[16 * 64];
  const int t = blockIdx.x & (N / 16 - 1);
  const int k = (blockIdx.x >> 8) & 3;
  const int b = blockIdx.x >> 10;
  const int n0 = t * 16;
  const float px = pos[(b * 4 + k) * 2 + 0];
  const float py = pos[(b * 4 + k) * 2 + 1];

  float lg[4], lb[4], bm[4], g0[4], g1[4], gb[4], qv[4];
  #pragma unroll
  for (int j = 0; j < 4; ++j){
    int col = ccol(j);
    lg[j] = ln_fg_g[col]; lb[j] = ln_fg_b[col];
    bm[j] = b_mlp_fg[col];
    g0[j] = w_grid[col * 4 + 0] - w_grid[col * 4 + 2];
    g1[j] = w_grid[col * 4 + 1] - w_grid[col * 4 + 3];
    gb[j] = b_grid[col];
    qv[j] = q_fg[(b * 4 + k) * 64 + col];
  }
  float rx[8], ry[8];
  #pragma unroll
  for (int r = 0; r < 8; ++r){
    int n = n0 + crow(r);
    rx[r] = (-1.f + (2.f / 63.f) * (float)(n & 63)) - px;
    ry[r] = (-1.f + (2.f / 63.f) * (float)(n >> 6)) - py;
  }

  v16h Bw[2][4]; load_WT(w_mlp_fg, Bw);
  v16h A[2]; v8f C[4];

  // ---- k path -> logits ----
  load_Ctile(kf + ((size_t)b * N + n0) * 64, C);
  #pragma unroll
  for (int r = 0; r < 8; ++r)
    #pragma unroll
    for (int j = 0; j < 4; ++j) C[j][r] += g0[j] * rx[r] + g1[j] * ry[r] + gb[j];
  ln_c_to_lds(C, lg, lb, S);
  __syncthreads();
  load_A(S, 64, A);
  v8f CK[4];
  #pragma unroll
  for (int j = 0; j < 4; ++j) CK[j] = vzero();
  mm16x64(A, Bw, CK);
  #pragma unroll
  for (int r = 0; r < 8; ++r){
    float p = 0.f;
    #pragma unroll
    for (int j = 0; j < 4; ++j) p += (CK[j][r] + bm[j]) * qv[j];
    p = rsum16(p) * SCALE;
    if ((threadIdx.x & 15) == 0)
      logits[((size_t)b * K + 1 + k) * N + n0 + crow(r)] = p;
  }
  __syncthreads();

  // ---- v path -> workspace ----
  load_Ctile(vf + ((size_t)b * N + n0) * 64, C);
  #pragma unroll
  for (int r = 0; r < 8; ++r)
    #pragma unroll
    for (int j = 0; j < 4; ++j) C[j][r] += g0[j] * rx[r] + g1[j] * ry[r] + gb[j];
  ln_c_to_lds(C, lg, lb, S);
  __syncthreads();
  load_A(S, 64, A);
  v8f CV[4];
  #pragma unroll
  for (int j = 0; j < 4; ++j) CV[j] = vzero();
  mm16x64(A, Bw, CV);
  #pragma unroll
  for (int r = 0; r < 8; ++r)
    #pragma unroll
    for (int j = 0; j < 4; ++j) CV[j][r] += bm[j];
  store_Ctile(CV, vout + (((size_t)b * 4 + k) * N + n0) * 64);
}

// ---------------------------------------------------------------------------
// Kernel 6 (per iter): a_bg[b,n] = (q_bg . k_bg[b,n]) * SCALE -> logits slot 0.
// ---------------------------------------------------------------------------
__global__ __launch_bounds__(256) void bg_logits(
    const float* __restrict__ kbg, const float* __restrict__ qbg,
    float* __restrict__ logits)
{
  using namespace sa;
  int wid  = (blockIdx.x * blockDim.x + threadIdx.x) >> 5;
  int nwav = (gridDim.x * blockDim.x) >> 5;
  int lane = threadIdx.x & 31;
  for (int row = wid; row < B * N; row += nwav){
    int b = row >> 12;
    const float* kr = kbg + (size_t)row * 64;
    const float* q  = qbg + b * 64;
    float p = kr[lane] * q[lane] + kr[lane + 32] * q[lane + 32];
    #pragma unroll
    for (int m = 1; m < 32; m <<= 1) p += __shfl_xor(p, m, 32);
    if (lane == 0) logits[(size_t)b * K * N + (row & (N - 1))] = p * SCALE;
  }
}

// ---------------------------------------------------------------------------
// Kernel 7 (per iter): softmax ACROSS SLOTS (axis K) + EPS; deterministic
// per-block partial slot sums (no float atomics -> graph-replay stable).
// ---------------------------------------------------------------------------
__global__ __launch_bounds__(256) void softmax_slots(
    const float* __restrict__ logits, float* __restrict__ attn,
    float* __restrict__ partials)
{
  using namespace sa;
  __shared__ float red[256];
  int b   = blockIdx.x >> 4;
  int blk = blockIdx.x & 15;
  int n   = blk * 256 + threadIdx.x;
  float l[K], e[K];
  float m = -1e30f;
  #pragma unroll
  for (int s = 0; s < K; ++s){ l[s] = logits[((size_t)b * K + s) * N + n]; m = fmaxf(m, l[s]); }
  float tot = 0.f;
  #pragma unroll
  for (int s = 0; s < K; ++s){ e[s] = expf(l[s] - m); tot += e[s]; }
  float itot = 1.f / tot;
  #pragma unroll
  for (int s = 0; s < K; ++s){
    e[s] = e[s] * itot + EPS;
    attn[((size_t)b * K + s) * N + n] = e[s];
  }
  for (int s = 0; s < K; ++s){
    red[threadIdx.x] = e[s];
    __syncthreads();
    for (int st = 128; st; st >>= 1){
      if (threadIdx.x < st) red[threadIdx.x] += red[threadIdx.x + st];
      __syncthreads();
    }
    if (threadIdx.x == 0) partials[(b * K + s) * 16 + blk] = red[0];
    __syncthreads();
  }
}

// ---------------------------------------------------------------------------
// Kernel 8 (per iter): normalize weights per slot over tokens (in place) and
// update fg positions: pos = clip(w@grid + tanh(w@w_pos^T + b_pos)/5, -1, 1).
// ---------------------------------------------------------------------------
__global__ __launch_bounds__(256) void norm_weights_pos(
    float* __restrict__ attn, const float* __restrict__ partials,
    const float* __restrict__ w_pos, const float* __restrict__ b_pos,
    float* __restrict__ pos)
{
  using namespace sa;
  __shared__ float red[256];
  __shared__ float res[4];
  int b = blockIdx.x / K, s = blockIdx.x % K;
  float tot = 0.f;
  for (int i = 0; i < 16; ++i) tot += partials[(b * K + s) * 16 + i];
  float inv = 1.f / tot;
  float* wrow = attn + ((size_t)b * K + s) * N;
  float ax = 0.f, ay = 0.f, qx = 0.f, qy = 0.f;
  for (int n = threadIdx.x; n < N; n += 256){
    float w = wrow[n] * inv;
    wrow[n] = w;
    float gx = -1.f + (2.f / 63.f) * (float)(n & 63);
    float gy = -1.f + (2.f / 63.f) * (float)(n >> 6);
    ax += w * gx; ay += w * gy;
    qx += w * w_pos[n]; qy += w * w_pos[N + n];
  }
  float vals[4] = {ax, ay, qx, qy};
  for (int i = 0; i < 4; ++i){
    red[threadIdx.x] = vals[i];
    __syncthreads();
    for (int st = 128; st; st >>= 1){
      if (threadIdx.x < st) red[threadIdx.x] += red[threadIdx.x + st];
      __syncthreads();
    }
    if (threadIdx.x == 0) res[i] = red[0];
    __syncthreads();
  }
  if (s > 0 && threadIdx.x == 0){
    int kk = s - 1;
    float npx = res[0] + tanhf(res[2] + b_pos[0]) * 0.2f;
    float npy = res[1] + tanhf(res[3] + b_pos[1]) * 0.2f;
    pos[(b * 4 + kk) * 2 + 0] = fminf(1.f, fmaxf(-1.f, npx));
    pos[(b * 4 + kk) * 2 + 1] = fminf(1.f, fmaxf(-1.f, npy));
  }
}

// ---------------------------------------------------------------------------
// Kernel 9 (iters 0..2): upd[b,s,:] = sum_n w[b,s,n] * v[b,s,n,:].
// ---------------------------------------------------------------------------
__global__ __launch_bounds__(256) void slot_update(
    const float* __restrict__ wattn, const float* __restrict__ vfg,
    const float* __restrict__ vbg, float* __restrict__ upd)
{
  using namespace sa;
  __shared__ float red[256];
  int b = blockIdx.x / K, s = blockIdx.x % K;
  int col = threadIdx.x & 63, grp = threadIdx.x >> 6;
  const float* wrow = wattn + ((size_t)b * K + s) * N;
  const float* vsrc = (s == 0) ? (vbg + (size_t)b * N * 64)
                               : (vfg + ((size_t)(b * 4 + s - 1)) * N * 64);
  float acc = 0.f;
  for (int n = grp; n < N; n += 4) acc += wrow[n] * vsrc[(size_t)n * 64 + col];
  red[threadIdx.x] = acc;
  __syncthreads();
  if (grp == 0)
    upd[(b * K + s) * 64 + col] = red[col] + red[64 + col] + red[128 + col] + red[192 + col];
}

// ---------------------------------------------------------------------------
// Kernel 10 (iters 0..2): GRU cell (r,z,n gates) + residual LN-MLP.
// ---------------------------------------------------------------------------
__global__ __launch_bounds__(64) void gru_residual(
    const float* __restrict__ upd,
    float* __restrict__ slot_fg, float* __restrict__ slot_bg,
    const float* __restrict__ fg_wih, const float* __restrict__ fg_whh,
    const float* __restrict__ fg_bih, const float* __restrict__ fg_bhh,
    const float* __restrict__ bg_wih, const float* __restrict__ bg_whh,
    const float* __restrict__ bg_bih, const float* __restrict__ bg_bhh,
    const float* __restrict__ rfg_g, const float* __restrict__ rfg_b,
    const float* __restrict__ w_rfg, const float* __restrict__ b_rfg,
    const float* __restrict__ rbg_g, const float* __restrict__ rbg_b,
    const float* __restrict__ w_rbg, const float* __restrict__ b_rbg)
{
  using namespace sa;
  __shared__ float shx[64], shh[64], shl[64];
  int b = blockIdx.x / K, s = blockIdx.x % K;
  bool bg = (s == 0);
  const float* x = upd + (b * K + s) * 64;
  float* h = bg ? slot_bg + b * 64 : slot_fg + (b * 4 + s - 1) * 64;
  const float* wih = bg ? bg_wih : fg_wih;
  const float* whh = bg ? bg_whh : fg_whh;
  const float* bih = bg ? bg_bih : fg_bih;
  const float* bhh = bg ? bg_bhh : fg_bhh;
  const float* lg  = bg ? rbg_g : rfg_g;
  const float* lbv = bg ? rbg_b : rfg_b;
  const float* wr  = bg ? w_rbg : w_rfg;
  const float* br  = bg ? b_rbg : b_rfg;
  int j = threadIdx.x;
  shx[j] = x[j]; shh[j] = h[j];
  __syncthreads();
  float gir = bih[j], giz = bih[64 + j], gin = bih[128 + j];
  float ghr = bhh[j], ghz = bhh[64 + j], ghn = bhh[128 + j];
  for (int d = 0; d < 64; ++d){
    float xv = shx[d], hv = shh[d];
    gir += xv * wih[j * 64 + d];
    giz += xv * wih[(64 + j) * 64 + d];
    gin += xv * wih[(128 + j) * 64 + d];
    ghr += hv * whh[j * 64 + d];
    ghz += hv * whh[(64 + j) * 64 + d];
    ghn += hv * whh[(128 + j) * 64 + d];
  }
  float r  = 1.f / (1.f + expf(-(gir + ghr)));
  float z  = 1.f / (1.f + expf(-(giz + ghz)));
  float nn = tanhf(gin + r * ghn);
  float hnew = (1.f - z) * nn + z * shh[j];
  __syncthreads();
  shx[j] = hnew;
  __syncthreads();
  float s1 = 0.f, s2 = 0.f;
  for (int d = 0; d < 64; ++d){ float v = shx[d]; s1 += v; s2 += v * v; }
  float mean = s1 * (1.f / 64.f);
  float var  = s2 * (1.f / 64.f) - mean * mean;
  float inv  = rsqrtf(var + 1e-5f);
  shl[j] = (hnew - mean) * inv * lg[j] + lbv[j];
  __syncthreads();
  float acc = br[j];
  for (int d = 0; d < 64; ++d) acc += shl[d] * wr[j * 64 + d];
  h[j] = hnew + acc;
}

// ---------------------------------------------------------------------------
// Kernel 11 (last iter): out[b,s] = [ sum_n w*v , sum_n w*fc_ln ]  (64 + 8).
// ---------------------------------------------------------------------------
__global__ __launch_bounds__(256) void final_out(
    const float* __restrict__ wattn, const float* __restrict__ vfg,
    const float* __restrict__ vbg, const float* __restrict__ fcln,
    float* __restrict__ out)
{
  using namespace sa;
  __shared__ float red[256];
  int b = blockIdx.x / K, s = blockIdx.x % K;
  int col = threadIdx.x & 63, grp = threadIdx.x >> 6;
  const float* wrow = wattn + ((size_t)b * K + s) * N;
  const float* vsrc = (s == 0) ? (vbg + (size_t)b * N * 64)
                               : (vfg + ((size_t)(b * 4 + s - 1)) * N * 64);
  float acc = 0.f;
  for (int n = grp; n < N; n += 4) acc += wrow[n] * vsrc[(size_t)n * 64 + col];
  red[threadIdx.x] = acc;
  __syncthreads();
  if (grp == 0)
    out[((size_t)b * K + s) * (SD + CD) + col] =
        red[col] + red[64 + col] + red[128 + col] + red[192 + col];
  __syncthreads();
  int c = threadIdx.x & 7, g2 = threadIdx.x >> 3;
  float ac = 0.f;
  for (int n = g2; n < N; n += 32) ac += wrow[n] * fcln[((size_t)b * N + n) * CD + c];
  red[threadIdx.x] = ac;
  __syncthreads();
  if (threadIdx.x < CD){
    float sres = 0.f;
    for (int g = 0; g < 32; ++g) sres += red[g * 8 + threadIdx.x];
    out[((size_t)b * K + s) * (SD + CD) + SD + threadIdx.x] = sres;
  }
}

// ---------------------------------------------------------------------------
// Host launcher.
// ---------------------------------------------------------------------------
extern "C" void kernel_launch(void* const* d_in, const int* in_sizes, int n_in,
                              void* d_out, int out_size, void* d_ws, size_t ws_size,
                              hipStream_t stream)
{
  using namespace sa;
  const float* feat      = (const float*)d_in[0];
  const float* featc     = (const float*)d_in[1];
  const float* w_grid    = (const float*)d_in[2];
  const float* b_grid    = (const float*)d_in[3];
  const float* w_kfg     = (const float*)d_in[4];
  const float* w_vfg     = (const float*)d_in[5];
  const float* w_kbg     = (const float*)d_in[6];
  const float* w_vbg     = (const float*)d_in[7];
  const float* ln_fg_g   = (const float*)d_in[8];
  const float* ln_fg_b   = (const float*)d_in[9];
  const float* w_mlp_fg  = (const float*)d_in[10];
  const float* b_mlp_fg  = (const float*)d_in[11];
  const float* ln_bg_g   = (const float*)d_in[12];
  const float* ln_bg_b   = (const float*)d_in[13];
  const float* w_mlp_bg  = (const float*)d_in[14];
  const float* b_mlp_bg  = (const float*)d_in[15];
  const float* slots0_fg = (const float*)d_in[16];
  const float* slots0_bg = (const float*)d_in[17];
  const float* fg_pos0   = (const float*)d_in[18];
  const float* w_pos     = (const float*)d_in[19];
  const float* b_pos     = (const float*)d_in[20];
  const float* ln_q_g    = (const float*)d_in[21];
  const float* ln_q_b    = (const float*)d_in[22];
  const float* w_q       = (const float*)d_in[23];
  const float* ln_qbg_g  = (const float*)d_in[24];
  const float* ln_qbg_b  = (const float*)d_in[25];
  const float* w_qbg     = (const float*)d_in[26];
  const float* gfg_wih   = (const float*)d_in[27];
  const float* gfg_whh   = (const float*)d_in[28];
  const float* gfg_bih   = (const float*)d_in[29];
  const float* gfg_bhh   = (const float*)d_in[30];
  const float* gbg_wih   = (const float*)d_in[31];
  const float* gbg_whh   = (const float*)d_in[32];
  const float* gbg_bih   = (const float*)d_in[33];
  const float* gbg_bhh   = (const float*)d_in[34];
  const float* ln_rfg_g  = (const float*)d_in[35];
  const float* ln_rfg_b  = (const float*)d_in[36];
  const float* w_rfg     = (const float*)d_in[37];
  const float* b_rfg     = (const float*)d_in[38];
  const float* ln_rbg_g  = (const float*)d_in[39];
  const float* ln_rbg_b  = (const float*)d_in[40];
  const float* w_rbg     = (const float*)d_in[41];
  const float* b_rbg     = (const float*)d_in[42];
  const float* ln_feat_g = (const float*)d_in[43];
  const float* ln_feat_b = (const float*)d_in[44];
  const float* ln_fc_g   = (const float*)d_in[45];
  const float* ln_fc_b   = (const float*)d_in[46];
  float* out = (float*)d_out;

  // Workspace layout (bytes, 256-aligned chunks). Total ~144 MB.
  char* ws = (char*)d_ws;
  size_t off = 0;
  auto take = [&](size_t bytes)->char*{
    char* p = ws + off;
    off = (off + bytes + 255) & ~(size_t)255;
    return p;
  };
  _Float16* featln = (_Float16*)take((size_t)B * N * D * sizeof(_Float16)); // 8 MB
  float* kf     = (float*)take((size_t)B * N * D * sizeof(float));          // 16 MB
  float* vf     = (float*)take((size_t)B * N * D * sizeof(float));          // 16 MB
  float* kbg    = (float*)take((size_t)B * N * SD * sizeof(float));         // 16 MB
  float* vbg    = (float*)take((size_t)B * N * SD * sizeof(float));         // 16 MB
  float* vfg    = (float*)take((size_t)B * 4 * N * SD * sizeof(float));     // 64 MB
  float* fcln   = (float*)take((size_t)B * N * CD * sizeof(float));         // 2 MB
  float* logits = (float*)take((size_t)B * K * N * sizeof(float));
  float* attn   = (float*)take((size_t)B * K * N * sizeof(float));
  float* parts  = (float*)take((size_t)B * K * 16 * sizeof(float));
  float* pos    = (float*)take((size_t)B * 4 * 2 * sizeof(float));
  float* slot_fg= (float*)take((size_t)B * 4 * SD * sizeof(float));
  float* slot_bg= (float*)take((size_t)B * SD * sizeof(float));
  float* q_fg   = (float*)take((size_t)B * 4 * SD * sizeof(float));
  float* q_bg   = (float*)take((size_t)B * SD * sizeof(float));
  float* upd    = (float*)take((size_t)B * K * SD * sizeof(float));
  (void)ws_size; (void)in_sizes; (void)n_in; (void)out_size;

  prep_ln<<<512, 256, 0, stream>>>(feat, ln_feat_g, ln_feat_b,
                                   featc, ln_fc_g, ln_fc_b, featln, fcln);
  proj_static<<<(B * N) / 16, 32, 0, stream>>>(featln, w_kfg, w_vfg, w_kbg, w_vbg,
                                               w_grid, b_grid, ln_bg_g, ln_bg_b,
                                               w_mlp_bg, b_mlp_bg, kf, vf, kbg, vbg);
  init_state<<<16, 256, 0, stream>>>(slots0_fg, slots0_bg, fg_pos0,
                                     slot_fg, slot_bg, pos);

  for (int it = 0; it < ITERS; ++it){
    q_proj<<<B * K, 64, 0, stream>>>(slot_fg, slot_bg, ln_q_g, ln_q_b, w_q,
                                     ln_qbg_g, ln_qbg_b, w_qbg, q_fg, q_bg);
    fg_kv_logits<<<B * 4 * (N / 16), 32, 0, stream>>>(kf, vf, pos, w_grid, b_grid,
                                                      ln_fg_g, ln_fg_b, w_mlp_fg,
                                                      b_mlp_fg, q_fg, vfg, logits);
    bg_logits<<<256, 256, 0, stream>>>(kbg, q_bg, logits);
    softmax_slots<<<B * 16, 256, 0, stream>>>(logits, attn, parts);
    norm_weights_pos<<<B * K, 256, 0, stream>>>(attn, parts, w_pos, b_pos, pos);
    if (it < ITERS - 1){
      slot_update<<<B * K, 256, 0, stream>>>(attn, vfg, vbg, upd);
      gru_residual<<<B * K, 64, 0, stream>>>(upd, slot_fg, slot_bg,
                                             gfg_wih, gfg_whh, gfg_bih, gfg_bhh,
                                             gbg_wih, gbg_whh, gbg_bih, gbg_bhh,
                                             ln_rfg_g, ln_rfg_b, w_rfg, b_rfg,
                                             ln_rbg_g, ln_rbg_b, w_rbg, b_rbg);
    } else {
      final_out<<<B * K, 256, 0, stream>>>(attn, vfg, vbg, fcln, out);
    }
  }
}
